// ConvolutionalSelfAttention_10050223472912
// MI455X (gfx1250) — matrix operand order repaired and checked by reading, not compiled, not run on hardware
//
#include <hip/hip_runtime.h>
#include <hip/hip_bf16.h>
#include <math.h>

// ---------------------------------------------------------------------------
// ConvolutionalSelfAttention for MI455X (gfx1250, wave32, WMMA)
//
// Pipeline (B=16, C=256, CA=128, tokens=256):
//  K0 prep      : Wq/Wk/Wv -> f16 transposed [CA][C]; Wo -> f16 transposed
//                 [C][CA]; mult[c] = 0.5*tanh(4*gamma+2.5)
//  K1 downsample: 8x bilinear decimation == 0.25 * 2x2 block at rows/cols
//                 {8i+3, 8i+4};  xs f16 [B][256][256]   (non-temporal reads)
//  K2 proj      : q,k row-major f16 [B][256][128];  v stored transposed
//                 v_t [B][128][256]     (v_wmma_f32_16x16x32_f16)
//  K3 scores    : s = q@k^T, fused row softmax -> attn f16 [B][256][256]
//  K4 attn@v    : o = attn @ v  (B-operand = v_t)   -> o f16 [B][256][128]
//  K5 out proj  : p = o @ Wo  (B-operand = wo_t)    -> p f32 [B][256][256]
//                 (Wo applied BEFORE upsample: linear ops commute, 64x fewer FLOPs)
//  K6 upsample  : out[b,h,w,c] = bilerp(p) * mult[c]  (256 MB non-temporal store)
//
// GEMM inner loops batch all B-fragment loads for a k-step into a register
// array before the WMMA burst so the backend can clause the loads and use
// staggered partial s_wait_loadcnt instead of a full drain per WMMA.
// ---------------------------------------------------------------------------

typedef __attribute__((ext_vector_type(16))) _Float16 v16h;
typedef __attribute__((ext_vector_type(8)))  _Float16 v8h;
typedef __attribute__((ext_vector_type(8)))  float    v8f;
typedef __attribute__((ext_vector_type(4)))  float    v4f;   // native vector: OK for nontemporal builtins

#define NB   16    // batches
#define NC   256   // input channels
#define NCA  128   // attention channels
#define NT   256   // tokens (16x16)

// ---- WMMA fragment helpers (ISA 7.12.2 layouts, wave32) --------------------

// A (16x32 f16, row-major source, ld in elements):
// lanes 0-15 : row m0+lane, K = {k0..k0+7} ++ {k0+16..k0+23}
// lanes 16-31: row m0+lane-16, K = {k0+8..k0+15} ++ {k0+24..k0+31}
__device__ __forceinline__ v16h frag_a(const _Float16* A, int ld, int m0, int k0, int lane) {
  const int rr = m0 + (lane & 15);
  const int kb = k0 + ((lane & 16) >> 1);     // +8 for upper half-wave
  const _Float16* p = A + rr * ld + kb;
  v8h lo = *(const v8h*)(p);
  v8h hi = *(const v8h*)(p + 16);
  return __builtin_shufflevector(lo, hi, 0,1,2,3,4,5,6,7,8,9,10,11,12,13,14,15);
}

// B (32x16 f16) supplied as row-major Bt = B^T (N x K):
// lanes 0-15 : col n0+lane, K = {k0..k0+7} ++ {k0+16..k0+23}
// lanes 16-31: col n0+lane-16, K = {k0+8..k0+15} ++ {k0+24..k0+31}
__device__ __forceinline__ v16h frag_b(const _Float16* Bt, int ld, int n0, int k0, int lane) {
  const int nn = n0 + (lane & 15);
  const int kk = k0 + ((lane & 16) >> 1);
  const _Float16* p = Bt + nn * ld + kk;
  v8h lo = *(const v8h*)(p);
  v8h hi = *(const v8h*)(p + 16);
  return __builtin_shufflevector(lo, hi, 0,1,2,3,4,5,6,7,8,9,10,11,12,13,14,15);
}

__device__ __forceinline__ v8f wmma_f16(v16h a, v16h b, v8f c) {
  return __builtin_amdgcn_wmma_f32_16x16x32_f16(false, a, false, b, (short)0, c, false, false);
}

// D (16x16 f32): VGPR r, lanes 0-15 -> (M=r, N=lane); lanes 16-31 -> (M=r+8)
__device__ __forceinline__ void store_d_f16(_Float16* D, int ld, int m0, int n0, int lane, v8f acc) {
  const int n  = n0 + (lane & 15);
  const int mb = m0 + ((lane >> 4) << 3);
#pragma unroll
  for (int r = 0; r < 8; ++r) D[(mb + r) * ld + n] = (_Float16)acc[r];
}

// transposed store: D^T[n][m], 8 consecutive halfs per lane (one b128 store)
__device__ __forceinline__ void store_d_f16_T(_Float16* Dt, int ldT, int m0, int n0, int lane, v8f acc) {
  const int n  = n0 + (lane & 15);
  const int mb = m0 + ((lane >> 4) << 3);
  _Float16* p = Dt + n * ldT + mb;
#pragma unroll
  for (int r = 0; r < 8; ++r) p[r] = (_Float16)acc[r];
}

// ---- K0: weight transpose/convert + multiplier -----------------------------
__global__ void __launch_bounds__(256)
k_prep(const float* Wq, const float* Wk, const float* Wv, const float* Wo,
       const float* gamma, _Float16* wq_t, _Float16* wk_t, _Float16* wv_t,
       _Float16* wo_t, float* mult) {
  int idx = blockIdx.x * 256 + threadIdx.x;
  const int NW = NC * NCA;  // 32768
  if (idx < 3 * NW) {
    int which = idx / NW, e = idx % NW;
    int d = e / NC, c = e % NC;                       // T[d][c] = W[c][d]
    const float* W = (which == 0) ? Wq : (which == 1) ? Wk : Wv;
    _Float16*    T = (which == 0) ? wq_t : (which == 1) ? wk_t : wv_t;
    T[e] = (_Float16)W[c * NCA + d];
  } else if (idx < 4 * NW) {
    int e = idx - 3 * NW;
    int c = e / NCA, d = e % NCA;                     // wo_t[c][d] = Wo[d][c]
    wo_t[e] = (_Float16)Wo[d * NC + c];
  } else if (idx < 4 * NW + NC) {
    int c = idx - 4 * NW;
    mult[c] = 0.5f * tanhf(4.0f * gamma[c] + 2.5f);
  }
}

// ---- K1: exact 8x bilinear downsample (avg of 2x2 at 8i+3,8i+4) ------------
__global__ void __launch_bounds__(256)
k_downsample(const float* __restrict__ x, _Float16* __restrict__ xs) {
  int b = blockIdx.x >> 8;
  int t = blockIdx.x & 255;
  int i = t >> 4, j = t & 15;
  int c = threadIdx.x;
  const float* xb = x + (((b * 128 + (8 * i + 3)) * 128) + (8 * j + 3)) * NC + c;
  float v = __builtin_nontemporal_load(xb) +
            __builtin_nontemporal_load(xb + NC) +
            __builtin_nontemporal_load(xb + 128 * NC) +
            __builtin_nontemporal_load(xb + 128 * NC + NC);
  xs[((b << 8) | t) * NC + c] = (_Float16)(0.25f * v);
}

// ---- K2: q/k/v projections (blockIdx.y: 0=q,1=k,2=v-transposed) ------------
__global__ void __launch_bounds__(256)
k_proj(const _Float16* __restrict__ xs, const _Float16* __restrict__ wq_t,
       const _Float16* __restrict__ wk_t, const _Float16* __restrict__ wv_t,
       _Float16* __restrict__ q, _Float16* __restrict__ k, _Float16* __restrict__ v_t) {
  int b = blockIdx.x, which = blockIdx.y;
  const _Float16* A  = xs + b * NT * NC;
  const _Float16* Bt = (which == 0) ? wq_t : (which == 1) ? wk_t : wv_t;
  int wave = threadIdx.x >> 5, lane = threadIdx.x & 31;
  int m0 = (wave + 8 * blockIdx.z) * 16;     // 16 M-tiles over 8 waves x gridz=2

  v8f acc[8] = {};
  for (int k0 = 0; k0 < NC; k0 += 32) {
    v16h a = frag_a(A, NC, m0, k0, lane);
    v16h bf[8];
#pragma unroll
    for (int nt = 0; nt < 8; ++nt) bf[nt] = frag_b(Bt, NC, nt * 16, k0, lane);
#pragma unroll
    for (int nt = 0; nt < 8; ++nt) acc[nt] = wmma_f16(a, bf[nt], acc[nt]);
  }
  if (which < 2) {
    _Float16* D = ((which == 0) ? q : k) + b * NT * NCA;
#pragma unroll
    for (int nt = 0; nt < 8; ++nt) store_d_f16(D, NCA, m0, nt * 16, lane, acc[nt]);
  } else {
    _Float16* Dt = v_t + b * NCA * NT;
#pragma unroll
    for (int nt = 0; nt < 8; ++nt) store_d_f16_T(Dt, NT, m0, nt * 16, lane, acc[nt]);
  }
}

// ---- K3: scores = q @ k^T, fused row-softmax -> attn f16 -------------------
__global__ void __launch_bounds__(256)
k_scores_softmax(const _Float16* __restrict__ q, const _Float16* __restrict__ kmat,
                 _Float16* __restrict__ attn) {
  int b = blockIdx.x;
  const _Float16* Aq = q    + b * NT * NCA;
  const _Float16* Bk = kmat + b * NT * NCA;   // k row-major == (k^T)^T : perfect Bt
  int wave = threadIdx.x >> 5, lane = threadIdx.x & 31;
  int m0 = (wave + 8 * blockIdx.y) * 16;

  v8f acc[16] = {};
  for (int k0 = 0; k0 < NCA; k0 += 32) {
    v16h a = frag_a(Aq, NCA, m0, k0, lane);
    v16h bf[16];
#pragma unroll
    for (int nt = 0; nt < 16; ++nt) bf[nt] = frag_b(Bk, NCA, nt * 16, k0, lane);
#pragma unroll
    for (int nt = 0; nt < 16; ++nt) acc[nt] = wmma_f16(a, bf[nt], acc[nt]);
  }

  const int nlane = lane & 15;
  const int mb    = m0 + ((lane >> 4) << 3);
  _Float16* Dst = attn + b * NT * NT;
#pragma unroll
  for (int r = 0; r < 8; ++r) {
    // row m = mb + r lives across lanes of this 16-lane half, over 16 n-tiles
    float mx = acc[0][r];
#pragma unroll
    for (int nt = 1; nt < 16; ++nt) mx = fmaxf(mx, acc[nt][r]);
    mx = fmaxf(mx, __shfl_xor(mx, 1, 32));
    mx = fmaxf(mx, __shfl_xor(mx, 2, 32));
    mx = fmaxf(mx, __shfl_xor(mx, 4, 32));
    mx = fmaxf(mx, __shfl_xor(mx, 8, 32));
    float e[16], s = 0.f;
#pragma unroll
    for (int nt = 0; nt < 16; ++nt) { e[nt] = __expf(acc[nt][r] - mx); s += e[nt]; }
    s += __shfl_xor(s, 1, 32);
    s += __shfl_xor(s, 2, 32);
    s += __shfl_xor(s, 4, 32);
    s += __shfl_xor(s, 8, 32);
    float inv = 1.0f / s;
    int row = mb + r;
#pragma unroll
    for (int nt = 0; nt < 16; ++nt)
      Dst[row * NT + nt * 16 + nlane] = (_Float16)(e[nt] * inv);
  }
}

// ---- K4: o = attn @ v  (Bt = v_t) ------------------------------------------
__global__ void __launch_bounds__(256)
k_attn_v(const _Float16* __restrict__ attn, const _Float16* __restrict__ v_t,
         _Float16* __restrict__ o) {
  int b = blockIdx.x;
  const _Float16* A  = attn + b * NT * NT;
  const _Float16* Bt = v_t  + b * NCA * NT;
  int wave = threadIdx.x >> 5, lane = threadIdx.x & 31;
  int m0 = (wave + 8 * blockIdx.y) * 16;

  v8f acc[8] = {};
  for (int k0 = 0; k0 < NT; k0 += 32) {
    v16h a = frag_a(A, NT, m0, k0, lane);
    v16h bf[8];
#pragma unroll
    for (int nt = 0; nt < 8; ++nt) bf[nt] = frag_b(Bt, NT, nt * 16, k0, lane);
#pragma unroll
    for (int nt = 0; nt < 8; ++nt) acc[nt] = wmma_f16(a, bf[nt], acc[nt]);
  }
  _Float16* D = o + b * NT * NCA;
#pragma unroll
  for (int nt = 0; nt < 8; ++nt) store_d_f16(D, NCA, m0, nt * 16, lane, acc[nt]);
}

// ---- K5: p = o @ Wo at 16x16 res (Wo commutes past upsample) ---------------
__global__ void __launch_bounds__(256)
k_out_proj(const _Float16* __restrict__ o, const _Float16* __restrict__ wo_t,
           float* __restrict__ p) {
  int b = blockIdx.x;
  const _Float16* A = o + b * NT * NCA;
  int wave = threadIdx.x >> 5, lane = threadIdx.x & 31;
  int m0 = (wave + 8 * blockIdx.y) * 16;

  v8f acc[16] = {};
  for (int k0 = 0; k0 < NCA; k0 += 32) {
    v16h a = frag_a(A, NCA, m0, k0, lane);
    v16h bf[16];
#pragma unroll
    for (int nt = 0; nt < 16; ++nt) bf[nt] = frag_b(wo_t, NCA, nt * 16, k0, lane);
#pragma unroll
    for (int nt = 0; nt < 16; ++nt) acc[nt] = wmma_f16(a, bf[nt], acc[nt]);
  }
  const int nlane = lane & 15;
  const int mb    = m0 + ((lane >> 4) << 3);
  float* Db = p + b * NT * NC;
#pragma unroll
  for (int nt = 0; nt < 16; ++nt)
#pragma unroll
    for (int r = 0; r < 8; ++r)
      Db[(mb + r) * NC + nt * 16 + nlane] = acc[nt][r];
}

// ---- K6: bilinear 8x upsample of p + per-channel multiplier (256 MB stream)-
__global__ void __launch_bounds__(256)
k_upsample(const float* __restrict__ p, const float* __restrict__ mult,
           float* __restrict__ out) {
  int t   = threadIdx.x;
  int pix = blockIdx.x * 4 + (t >> 6);
  int b   = pix >> 14;
  int hw  = pix & 16383;
  int h   = hw >> 7, w = hw & 127;
  int cg  = (t & 63) << 2;

  float lh = h * 0.125f - 0.4375f;
  float fh = floorf(lh);
  float ah = lh - fh;
  int i0 = (int)fh, i1 = i0 + 1;
  i0 = max(i0, 0); i1 = min(i1, 15);

  float lw = w * 0.125f - 0.4375f;
  float fw = floorf(lw);
  float aw = lw - fw;
  int j0 = (int)fw, j1 = j0 + 1;
  j0 = max(j0, 0); j1 = min(j1, 15);

  const float* pb = p + b * NT * NC;
  v4f v00 = *(const v4f*)(pb + (i0 * 16 + j0) * NC + cg);
  v4f v01 = *(const v4f*)(pb + (i0 * 16 + j1) * NC + cg);
  v4f v10 = *(const v4f*)(pb + (i1 * 16 + j0) * NC + cg);
  v4f v11 = *(const v4f*)(pb + (i1 * 16 + j1) * NC + cg);
  float w00 = (1.f - ah) * (1.f - aw), w01 = (1.f - ah) * aw;
  float w10 = ah * (1.f - aw),         w11 = ah * aw;
  v4f m = *(const v4f*)(mult + cg);
  v4f r;
  r.x = (v00.x * w00 + v01.x * w01 + v10.x * w10 + v11.x * w11) * m.x;
  r.y = (v00.y * w00 + v01.y * w01 + v10.y * w10 + v11.y * w11) * m.y;
  r.z = (v00.z * w00 + v01.z * w01 + v10.z * w10 + v11.z * w11) * m.z;
  r.w = (v00.w * w00 + v01.w * w01 + v10.w * w10 + v11.w * w11) * m.w;
  // write-once 256 MB stream: non-temporal so it doesn't evict L2-resident
  // intermediates (p, weights) or linger in cache
  __builtin_nontemporal_store(r, (v4f*)(out + ((b * 128 + h) * 128 + w) * NC + cg));
}

// ---------------------------------------------------------------------------
extern "C" void kernel_launch(void* const* d_in, const int* in_sizes, int n_in,
                              void* d_out, int out_size, void* d_ws, size_t ws_size,
                              hipStream_t stream) {
  const float* x     = (const float*)d_in[0];
  const float* Wq    = (const float*)d_in[1];
  const float* Wk    = (const float*)d_in[2];
  const float* Wv    = (const float*)d_in[3];
  const float* Wo    = (const float*)d_in[4];
  const float* gamma = (const float*)d_in[5];
  float* out = (float*)d_out;

  char* w = (char*)d_ws;
  auto carve = [&](size_t bytes) -> char* {
    char* r = w;
    w += (bytes + 255) & ~(size_t)255;
    return r;
  };
  _Float16* xs   = (_Float16*)carve((size_t)NB * NT * NC  * 2);  // 2 MB
  _Float16* wq_t = (_Float16*)carve((size_t)NCA * NC * 2);
  _Float16* wk_t = (_Float16*)carve((size_t)NCA * NC * 2);
  _Float16* wv_t = (_Float16*)carve((size_t)NCA * NC * 2);
  _Float16* wo_t = (_Float16*)carve((size_t)NC * NCA * 2);
  float*    mult = (float*)   carve((size_t)NC * 4);
  _Float16* q    = (_Float16*)carve((size_t)NB * NT * NCA * 2);  // 1 MB
  _Float16* kbuf = (_Float16*)carve((size_t)NB * NT * NCA * 2);
  _Float16* v_t  = (_Float16*)carve((size_t)NB * NCA * NT * 2);
  _Float16* attn = (_Float16*)carve((size_t)NB * NT * NT  * 2);  // 2 MB
  _Float16* o    = (_Float16*)carve((size_t)NB * NT * NCA * 2);
  float*    p    = (float*)   carve((size_t)NB * NT * NC  * 4);  // 4 MB
  (void)ws_size; (void)in_sizes; (void)n_in; (void)out_size;

  k_prep<<<dim3(513), dim3(256), 0, stream>>>(Wq, Wk, Wv, Wo, gamma,
                                              wq_t, wk_t, wv_t, wo_t, mult);
  k_downsample<<<dim3(NB * NT), dim3(256), 0, stream>>>(x, xs);
  k_proj<<<dim3(NB, 3, 2), dim3(256), 0, stream>>>(xs, wq_t, wk_t, wv_t, q, kbuf, v_t);
  k_scores_softmax<<<dim3(NB, 2), dim3(256), 0, stream>>>(q, kbuf, attn);
  k_attn_v<<<dim3(NB, 2), dim3(256), 0, stream>>>(attn, v_t, o);
  k_out_proj<<<dim3(NB, 2), dim3(256), 0, stream>>>(o, wo_t, p);
  k_upsample<<<dim3(NB * 128 * 128 / 4), dim3(256), 0, stream>>>(p, mult, out);
}
